// Stagenet_58239756533970
// MI455X (gfx1250) — matrix operand order repaired and checked by reading, not compiled, not run on hardware
//
#include <hip/hip_runtime.h>
#include <stdint.h>
#include <stddef.h>

// ---- problem constants ----
#define T_   256
#define B_   256
#define F_   128
#define H_   384
#define L_   3
#define KSE  10           // ring buffer depth K
#define C_   128          // H/L
#define G_   1542         // 4H + 2L
#define S_   64           // H/6
#define GP   1600         // G padded to 100*16
#define NT   100          // GP/16 column tiles
#define NG   25           // groups of 4 N-tiles
#define KTOT 512          // F + H (bias folded out)
#define KT   16           // KTOT/32 k-tiles

typedef __bf16 bf16_t;
typedef bf16_t v16bf __attribute__((ext_vector_type(16)));
typedef bf16_t v8bf  __attribute__((ext_vector_type(8)));
typedef float  v8f   __attribute__((ext_vector_type(8)));

__device__ __forceinline__ unsigned short f2bf(float f) {
  uint32_t u = __builtin_bit_cast(uint32_t, f);
  uint32_t lsb = (u >> 16) & 1u;
  u += 0x7FFFu + lsb;                    // round-to-nearest-even
  return (unsigned short)(u >> 16);
}
__device__ __forceinline__ float sigmoidf_(float x) { return 1.0f / (1.0f + __expf(-x)); }

// ---------------- one-time packing ----------------

// Pack [Wk(0:128); Wr(0:384)] -> bf16 blocks in WMMA B-fragment order,
// grouped so each N-group's 64KB slice is contiguous:
//   element index = ((ng*16 + kt)*4 + ni)*512 + lane*16 + j
//   with lane = n + 16*((kk>>3)&1), j = (kk & 7) + 8*(kk>>4), nt = ng*4 + ni.
__global__ __launch_bounds__(256) void pack_w(const float* __restrict__ Wk,
                                              const float* __restrict__ Wr,
                                              unsigned short* __restrict__ Wpack) {
  int idx = blockIdx.x * 256 + threadIdx.x;        // 0 .. NG*KT*4*512-1
  int within = idx & 511;
  int blk    = idx >> 9;
  int lane = within >> 4;
  int j    = within & 15;
  int ni = blk & 3;
  int kt = (blk >> 2) & 15;
  int ng = blk >> 6;
  int kk = (j & 7) + 8 * (lane >> 4) + 16 * (j >> 3);
  int n  = lane & 15;
  int kg = kt * 32 + kk;
  int ngl = (ng * 4 + ni) * 16 + n;
  float v = 0.0f;
  if (ngl < G_) v = (kg < F_) ? Wk[(size_t)kg * G_ + ngl] : Wr[(size_t)(kg - F_) * G_ + ngl];
  Wpack[idx] = f2bf(v);
}

// biasG[g] = bk+br+Wk[128,:]+Wr[384,:] (the folded "1" rows)
__global__ __launch_bounds__(256) void pack_bias(const float* __restrict__ Wk,
                                                 const float* __restrict__ bk,
                                                 const float* __restrict__ Wr,
                                                 const float* __restrict__ br,
                                                 float* __restrict__ biasG) {
  int g = blockIdx.x * 256 + threadIdx.x;
  if (g >= GP) return;
  float v = 0.0f;
  if (g < G_) v = bk[g] + br[g] + Wk[(size_t)F_ * G_ + g] + Wr[(size_t)H_ * G_ + g];
  biasG[g] = v;
}

__global__ __launch_bounds__(256) void pack_x(const float* __restrict__ x,
                                              unsigned short* __restrict__ Xbf) {
  size_t i = (size_t)blockIdx.x * 256 + threadIdx.x;   // T*B*F threads exactly
  Xbf[i] = f2bf(x[i]);
}

__global__ __launch_bounds__(256) void zero_f32(float* __restrict__ p, size_t n) {
  size_t i = (size_t)blockIdx.x * blockDim.x + threadIdx.x;
  size_t stride = (size_t)gridDim.x * blockDim.x;
  for (; i < n; i += stride) p[i] = 0.0f;
}

// ---------------- per-step recurrent GEMM (WMMA bf16 + async LDS staging) ----------------
// xo[b, 0:GP] = [x_t | h] @ Wcat + bias.
// Grid: 50 blocks = 25 N-groups x 2 M-halves; 8 waves/block; wave = 1 M-tile x 4 N-tiles.
// The block's 64KB weight slice is staged into LDS with async loads (ASYNCcnt),
// then consumed via ds_load_b128 B-fragments.
__global__ __launch_bounds__(256) void gemm_step(const unsigned short* __restrict__ Xbf,
                                                 const unsigned short* __restrict__ Hbf,
                                                 const unsigned short* __restrict__ Wpack,
                                                 const float* __restrict__ biasG,
                                                 float* __restrict__ xo, int t) {
  __shared__ unsigned short Bsm[KT * 4 * 512];           // 64 KB
  const int lane  = threadIdx.x & 31;
  const int wv    = threadIdx.x >> 5;                    // 0..7
  const int ng    = blockIdx.x >> 1;                     // N-group 0..24
  const int mhalf = blockIdx.x & 1;                      // 0..1

  // ---- stage this block's weight slice: global -> LDS, async (16B/lane/iter) ----
  {
    const char* gsrc = (const char*)(Wpack + (size_t)ng * (KT * 4 * 512));
    unsigned lbase = (unsigned)(uintptr_t)(&Bsm[0]);     // low 32 bits = LDS byte offset
#pragma unroll
    for (int it = 0; it < 16; ++it) {
      unsigned byteoff = (unsigned)(it * 256 + threadIdx.x) * 16u;
      unsigned ldsaddr = lbase + byteoff;
      const char* g = gsrc + byteoff;
      asm volatile("global_load_async_to_lds_b128 %0, %1, off"
                   :: "v"(ldsaddr), "v"(g) : "memory");
    }
    asm volatile("s_wait_asynccnt 0x0" ::: "memory");
  }
  __syncthreads();

  // ---- per-wave tile ----
  const int mt  = mhalf * 8 + wv;                        // M-tile 0..15
  const int hi  = lane >> 4;
  const int row = mt * 16 + (lane & 15);
  const unsigned short* xrow = Xbf + ((size_t)t * B_ + row) * F_ + hi * 8;
  const unsigned short* hrow = Hbf + (size_t)row * H_ + hi * 8;

  v8f acc[4];
#pragma unroll
  for (int ni = 0; ni < 4; ++ni) {
    float bv = biasG[(ng * 4 + ni) * 16 + (lane & 15)];
#pragma unroll
    for (int i = 0; i < 8; ++i) acc[ni][i] = bv;
  }

#pragma unroll
  for (int kt = 0; kt < KT; ++kt) {
    // A fragment: ISA 16-bit A layout; chunk0 K = kt*32 + hi*8, chunk1 = +16
    const unsigned short* s = (kt < 4) ? (xrow + kt * 32) : (hrow + (kt * 32 - F_));
    v8bf a_lo = *(const v8bf*)(s);
    v8bf a_hi = *(const v8bf*)(s + 16);
    v16bf a = __builtin_shufflevector(a_lo, a_hi, 0,1,2,3,4,5,6,7,8,9,10,11,12,13,14,15);
#pragma unroll
    for (int ni = 0; ni < 4; ++ni) {                     // 4 independent accumulators
      v16bf bm = *(const v16bf*)(&Bsm[(kt * 4 + ni) * 512 + lane * 16]);
      acc[ni] = __builtin_amdgcn_wmma_f32_16x16x32_bf16(false, a, false, bm,
                                                        (short)0, acc[ni], false, false);
    }
  }

  // ---- store C (layout: M = v + 8*hi, N = lane&15) ----
#pragma unroll
  for (int ni = 0; ni < 4; ++ni) {
    const int nc = (ng * 4 + ni) * 16 + (lane & 15);
#pragma unroll
    for (int v = 0; v < 8; ++v) {
      const int m = mt * 16 + v + 8 * hi;
      xo[(size_t)m * GP + nc] = acc[ni][v];
    }
  }
}

// ---------------- per-step cell (gates, state, ring buffers) ----------------
__global__ __launch_bounds__(256) void cell_step(const float* __restrict__ xo,
                                                 float* __restrict__ cbuf,
                                                 float* __restrict__ hbuf,
                                                 unsigned short* __restrict__ Hbf,
                                                 float* __restrict__ buf_h,
                                                 float* __restrict__ buf_d, int t) {
  int tid = blockIdx.x * 256 + threadIdx.x;              // B*C threads
  int b = tid >> 7;           // / C_
  int j = tid & 127;          // % C_
  const float* xr = xo + (size_t)b * GP;

  // master forget gate: fm = cumsum(softmax(xo[:3]))
  float a0 = xr[0], a1 = xr[1], a2 = xr[2];
  float m = fmaxf(a0, fmaxf(a1, a2));
  float e0 = __expf(a0 - m), e1 = __expf(a1 - m), e2 = __expf(a2 - m);
  float inv = 1.0f / (e0 + e1 + e2);
  float p0 = e0 * inv, p1 = e1 * inv, p2 = e2 * inv;
  float fm[3] = { p0, p0 + p1, p0 + p1 + p2 };

  // master input gate: im[l] = suffix-sum of softmax(xo[3:6])
  float c0 = xr[3], c1 = xr[4], c2 = xr[5];
  float mm = fmaxf(c0, fmaxf(c1, c2));
  float f0 = __expf(c0 - mm), f1 = __expf(c1 - mm), f2 = __expf(c2 - mm);
  float inv2 = 1.0f / (f0 + f1 + f2);
  float q0 = f0 * inv2, q1 = f1 * inv2, q2 = f2 * inv2;
  float im[3] = { q0 + q1 + q2, q1 + q2, q2 };

  const int slot = t % KSE;
#pragma unroll
  for (int l = 0; l < L_; ++l) {
    float fg  = sigmoidf_(xr[6 + (0 * L_ + l) * C_ + j]);
    float ig  = sigmoidf_(xr[6 + (1 * L_ + l) * C_ + j]);
    float og  = sigmoidf_(xr[6 + (2 * L_ + l) * C_ + j]);
    float cin = tanhf    (xr[6 + (3 * L_ + l) * C_ + j]);
    size_t hidx = (size_t)b * H_ + l * C_ + j;
    float cl = cbuf[hidx];
    float ov = fm[l] * im[l];
    float cn = ov * (fg * cl + ig * cin) + (fm[l] - ov) * cl + (im[l] - ov) * cin;
    float hn = og * tanhf(cn);
    cbuf[hidx] = cn;
    hbuf[hidx] = hn;
    Hbf[hidx]  = f2bf(hn);                               // next step's A-matrix (bf16)
    buf_h[(size_t)slot * B_ * H_ + hidx] = hn;
  }
  if (j == 0) buf_d[slot * B_ + b] = 1.0f - (fm[0] + fm[1] + fm[2]) * (1.0f / 3.0f);
}

// ---------------- final-step locality head (one-time) ----------------

__global__ __launch_bounds__(256) void loc_dis(const float* __restrict__ buf_d,
                                               float* __restrict__ dis) {
  int b = threadIdx.x;                                   // 256 threads
  float cum[KSE]; float run = 0.0f;
#pragma unroll
  for (int k = 0; k < KSE; ++k) {                        // time order: slot (T-10+k)%10 = (6+k)%10
    run += buf_d[((6 + k) % KSE) * B_ + b];
    cum[k] = run;
  }
  float m = cum[0];
#pragma unroll
  for (int k = 1; k < KSE; ++k) m = fmaxf(m, cum[k]);
  float s = 0.0f; float e[KSE];
#pragma unroll
  for (int k = 0; k < KSE; ++k) { e[k] = __expf(cum[k] - m); s += e[k]; }
  float invs = 1.0f / s;
#pragma unroll
  for (int k = 0; k < KSE; ++k) dis[b * KSE + k] = e[k] * invs;
}

// local_h -> A2[b, k*H+i]; mh = mean_k local_h
__global__ __launch_bounds__(256) void loc_localh(const float* __restrict__ buf_h,
                                                  const float* __restrict__ dis,
                                                  float* __restrict__ A2,
                                                  float* __restrict__ mh) {
  int tid = blockIdx.x * 256 + threadIdx.x;              // B*H threads
  int b = tid / H_, i = tid % H_;
  float acc = 0.0f;
#pragma unroll
  for (int k = 0; k < KSE; ++k) {
    int slot = (6 + k) % KSE;
    float v = buf_h[(size_t)slot * B_ * H_ + (size_t)b * H_ + i] * dis[b * KSE + k];
    A2[(size_t)b * (KSE * H_) + k * H_ + i] = v;
    acc += v;
  }
  mh[(size_t)b * H_ + i] = acc * (1.0f / KSE);
}

__global__ __launch_bounds__(256) void loc_mlp1(const float* __restrict__ mh,
                                                const float* __restrict__ Ws,
                                                const float* __restrict__ bs,
                                                float* __restrict__ t1) {
  int tid = blockIdx.x * 256 + threadIdx.x;              // B*S threads
  int b = tid / S_, s = tid % S_;
  float acc = bs[s];
  const float* mr = mh + (size_t)b * H_;
  for (int i = 0; i < H_; ++i) acc += mr[i] * Ws[(size_t)i * S_ + s];
  t1[tid] = fmaxf(acc, 0.0f);
}

__global__ __launch_bounds__(256) void loc_mlp2(const float* __restrict__ t1,
                                                const float* __restrict__ Wrs,
                                                const float* __restrict__ brs,
                                                float* __restrict__ theme) {
  int tid = blockIdx.x * 256 + threadIdx.x;              // B*H threads
  int b = tid / H_, i = tid % H_;
  float acc = brs[i];
  const float* tr = t1 + (size_t)b * S_;
  for (int s = 0; s < S_; ++s) acc += tr[s] * Wrs[(size_t)s * H_ + i];
  theme[tid] = sigmoidf_(acc);
}

// conv[b,o] = sum_{k,i} A2[b,k,i]*Wc[o,i,k] + bc[o];  rnn = theme*conv + h_last
__global__ __launch_bounds__(256) void loc_conv(const float* __restrict__ A2,
                                                const float* __restrict__ Wc,
                                                const float* __restrict__ bc,
                                                const float* __restrict__ theme,
                                                const float* __restrict__ hbuf,
                                                float* __restrict__ rnn) {
  int tid = blockIdx.x * 256 + threadIdx.x;              // B*H threads
  int b = tid / H_, o = tid % H_;
  float acc = bc[o];
  const float* a2 = A2 + (size_t)b * (KSE * H_);
  const float* wc = Wc + (size_t)o * H_ * KSE;
  for (int k = 0; k < KSE; ++k) {
    const float* a2k = a2 + k * H_;
    for (int i = 0; i < H_; ++i) acc += a2k[i] * wc[(size_t)i * KSE + k];
  }
  rnn[tid] = theme[tid] * acc + hbuf[tid];
}

__global__ __launch_bounds__(256) void out_head(const float* __restrict__ rnn,
                                                const float* __restrict__ Wo,
                                                const float* __restrict__ bo,
                                                float* __restrict__ out) {
  int b = threadIdx.x;                                   // 256 threads
  float acc = bo[0];
  const float* r = rnn + (size_t)b * H_;
  for (int o = 0; o < H_; ++o) acc += r[o] * Wo[o];
  out[b] = sigmoidf_(acc);
}

// ---------------- host orchestration ----------------

extern "C" void kernel_launch(void* const* d_in, const int* in_sizes, int n_in,
                              void* d_out, int out_size, void* d_ws, size_t ws_size,
                              hipStream_t stream) {
  const float* x   = (const float*)d_in[0];
  const float* Wk  = (const float*)d_in[1];
  const float* bk  = (const float*)d_in[2];
  const float* Wr  = (const float*)d_in[3];
  const float* br  = (const float*)d_in[4];
  const float* Ws  = (const float*)d_in[5];
  const float* bs  = (const float*)d_in[6];
  const float* Wrs = (const float*)d_in[7];
  const float* brs = (const float*)d_in[8];
  const float* Wc  = (const float*)d_in[9];
  const float* bc  = (const float*)d_in[10];
  const float* Wo  = (const float*)d_in[11];
  const float* bo  = (const float*)d_in[12];
  float* out = (float*)d_out;

  char* wsb = (char*)d_ws;
  size_t off = 0;
  auto alloc = [&](size_t bytes) -> char* {
    char* p = wsb + off;
    off = (off + bytes + 255) & ~(size_t)255;
    return p;
  };

  unsigned short* Wpack = (unsigned short*)alloc((size_t)NG * KT * 4 * 512 * 2); // 1.6 MB
  float*          biasG = (float*)alloc((size_t)GP * 4);
  unsigned short* Xbf   = (unsigned short*)alloc((size_t)T_ * B_ * F_ * 2);      // 16.8 MB
  float*          xo    = (float*)alloc((size_t)B_ * GP * 4);                    // 1.6 MB
  // zero-initialized recurrent state, laid out contiguously:
  size_t zr_floats = (size_t)B_ * H_            // c
                   + (size_t)KSE * B_ * H_      // buf_h
                   + (size_t)KSE * B_           // buf_d
                   + (size_t)B_ * H_ / 2;       // Hbf (bf16, B*H ushorts)
  float* zbase  = (float*)alloc(zr_floats * 4);
  float* cbuf   = zbase;
  float* buf_h  = cbuf + (size_t)B_ * H_;
  float* buf_d  = buf_h + (size_t)KSE * B_ * H_;
  unsigned short* Hbf = (unsigned short*)(buf_d + (size_t)KSE * B_);
  float* hbuf   = (float*)alloc((size_t)B_ * H_ * 4);
  float* dis    = (float*)alloc((size_t)B_ * KSE * 4);
  float* mh     = (float*)alloc((size_t)B_ * H_ * 4);
  float* A2     = (float*)alloc((size_t)B_ * KSE * H_ * 4);
  float* t1v    = (float*)alloc((size_t)B_ * S_ * 4);
  float* theme  = (float*)alloc((size_t)B_ * H_ * 4);
  float* rnn    = (float*)alloc((size_t)B_ * H_ * 4);

  // one-time packs + state zeroing
  pack_w  <<<(NG * KT * 4 * 512) / 256, 256, 0, stream>>>(Wk, Wr, Wpack);
  pack_bias<<<(GP + 255) / 256,          256, 0, stream>>>(Wk, bk, Wr, br, biasG);
  pack_x  <<<(T_ * B_ * F_) / 256,       256, 0, stream>>>(x, Xbf);
  zero_f32<<<1024, 256, 0, stream>>>(zbase, zr_floats);

  // sequential recurrence: 256 steps, 2 kernels each (ordering via stream)
  for (int t = 0; t < T_; ++t) {
    gemm_step<<<2 * NG, 256, 0, stream>>>(Xbf, Hbf, Wpack, biasG, xo, t);
    cell_step<<<(B_ * C_) / 256, 256, 0, stream>>>(xo, cbuf, hbuf, Hbf, buf_h, buf_d, t);
  }

  // final-step locality head (only loc_all[-1] is used by the reference)
  loc_dis   <<<1, 256, 0, stream>>>(buf_d, dis);
  loc_localh<<<(B_ * H_) / 256, 256, 0, stream>>>(buf_h, dis, A2, mh);
  loc_mlp1  <<<(B_ * S_) / 256, 256, 0, stream>>>(mh, Ws, bs, t1v);
  loc_mlp2  <<<(B_ * H_) / 256, 256, 0, stream>>>(t1v, Wrs, brs, theme);
  loc_conv  <<<(B_ * H_) / 256, 256, 0, stream>>>(A2, Wc, bc, theme, hbuf, rnn);
  out_head  <<<1, 256, 0, stream>>>(rnn, Wo, bo, out);
}
